// BiLSTMCRF_12532714569927
// MI455X (gfx1250) — compile-verified, matrix operand order checked by reading, not run on hardware
//
#include <hip/hip_runtime.h>

#define SEQLEN   8192
#define EDIM     384
#define HDIM     192
#define GDIM     768           // 4*H
#define NTAG     11
#define START_TAG 9
#define STOP_TAG  10
#define NEGV     (-10000.0f)
#define WPAD     196           // padded bf16 row stride for w_hh in LDS (bank-conflict-free)

typedef __bf16 bf16_t;
typedef bf16_t v16bf __attribute__((ext_vector_type(16)));
typedef bf16_t v4bf  __attribute__((ext_vector_type(4)));
typedef float  v8f   __attribute__((ext_vector_type(8)));
typedef unsigned int u32x4 __attribute__((ext_vector_type(4)));

union FragAB { u32x4 q[2]; v16bf v; };

// ---------------------------------------------------------------------------
// 1) Embedding gather + f32 -> bf16 convert:  xbf[t][e] = bf16(emb[sent[t]][e])
// ---------------------------------------------------------------------------
__global__ void embed_bf16_kernel(const int* __restrict__ sent,
                                  const float* __restrict__ emb,
                                  bf16_t* __restrict__ xbf) {
#if defined(__HIP_DEVICE_COMPILE__)
    int idx = blockIdx.x * 256 + threadIdx.x;
    if (idx >= SEQLEN * EDIM) return;
    int t = idx / EDIM;
    int e = idx - t * EDIM;
    xbf[idx] = (bf16_t)emb[(long)sent[t] * EDIM + e];
#endif
}

// ---------------------------------------------------------------------------
// 2) Generic f32 -> bf16 convert (for w_ih_f / w_ih_b)
// ---------------------------------------------------------------------------
__global__ void cvt_bf16_kernel(const float* __restrict__ in,
                                bf16_t* __restrict__ out, int n) {
#if defined(__HIP_DEVICE_COMPILE__)
    int idx = blockIdx.x * 256 + threadIdx.x;
    if (idx < n) out[idx] = (bf16_t)in[idx];
#endif
}

// ---------------------------------------------------------------------------
// 3) Input-projection GEMM via WMMA bf16 -> f32:
//    G[t][j] = sum_e x[t][e] * w_ih[j][e]  + (b_ih[j] + b_hh[j])
//    One wave per 16x16 output tile; K swept 32 at a time with
//    v_wmma_f32_16x16x32_bf16. Fragments loaded as two b128 per operand/lane
//    following the ISA VGPR layout (M/N = lane&15, K half selected by lane>=16).
// ---------------------------------------------------------------------------
__global__ void __launch_bounds__(256)
gates_gemm_kernel(const bf16_t* __restrict__ xbf,
                  const bf16_t* __restrict__ wbf,
                  const float* __restrict__ b_ih,
                  const float* __restrict__ b_hh,
                  float* __restrict__ G) {
#if defined(__HIP_DEVICE_COMPILE__)
    const int wave = threadIdx.x >> 5;
    const int lane = threadIdx.x & 31;
    const int tile = blockIdx.x * 8 + wave;        // 512*48 = 24576 tiles
    const int mt = tile / (GDIM / 16);
    const int nt = tile - mt * (GDIM / 16);

    const int r  = lane & 15;                      // A: row M; B: col N
    const int kh = (lane >> 4) << 3;               // K sub-half: 0 or 8

    const bf16_t* arow = xbf + (size_t)(mt * 16 + r) * EDIM;
    const bf16_t* brow = wbf + (size_t)(nt * 16 + r) * EDIM;

    v8f c = {};
    #pragma unroll
    for (int k0 = 0; k0 < EDIM; k0 += 32) {
        FragAB a, b;
        a.q[0] = *reinterpret_cast<const u32x4*>(arow + k0 + kh);        // K = kh..kh+7
        a.q[1] = *reinterpret_cast<const u32x4*>(arow + k0 + 16 + kh);   // K = 16+kh..
        b.q[0] = *reinterpret_cast<const u32x4*>(brow + k0 + kh);
        b.q[1] = *reinterpret_cast<const u32x4*>(brow + k0 + 16 + kh);
        c = __builtin_amdgcn_wmma_f32_16x16x32_bf16(
                false, a.v, false, b.v, (short)0, c, false, false);
    }

    // C/D layout: VGPR rr -> M = rr + 8*(lane>=16), N = lane&15
    const int n  = lane & 15;
    const int j  = nt * 16 + n;
    const float bias = b_ih[j] + b_hh[j];
    const int mbase = mt * 16 + ((lane >> 4) << 3);
    #pragma unroll
    for (int rr = 0; rr < 8; ++rr)
        G[(size_t)(mbase + rr) * GDIM + j] = c[rr] + bias;
#endif
}

// ---------------------------------------------------------------------------
// 4) Persistent bidirectional LSTM scan. blockIdx.x = direction.
//    w_hh cached in 320KB LDS as bf16 (row stride WPAD=196 elems to spread
//    banks); thread j owns gate output j; h/c/gate vectors live in LDS.
// ---------------------------------------------------------------------------
__global__ void __launch_bounds__(GDIM)
lstm_scan_kernel(const float* __restrict__ whh_f, const float* __restrict__ whh_b,
                 const float* __restrict__ Gf,    const float* __restrict__ Gb,
                 float* __restrict__ hf,          float* __restrict__ hb) {
#if defined(__HIP_DEVICE_COMPILE__)
    extern __shared__ char smem[];
    bf16_t* wsm = (bf16_t*)smem;                       // GDIM * WPAD bf16
    float*  h_s = (float*)(smem + GDIM * WPAD * 2);    // HDIM
    float*  c_s = h_s + HDIM;                          // HDIM
    float*  g_s = c_s + HDIM;                          // GDIM

    const int dir = blockIdx.x;
    const float* whh  = dir ? whh_b : whh_f;
    const float* G    = dir ? Gb : Gf;
    float*       hout = dir ? hb : hf;
    const int j = threadIdx.x;

    // one-time: stage w_hh into LDS as bf16 (coalesced global reads)
    for (int i = j; i < GDIM * HDIM; i += GDIM) {
        int row = i / HDIM, col = i - row * HDIM;
        wsm[row * WPAD + col] = (bf16_t)whh[i];
    }
    if (j < HDIM) { h_s[j] = 0.f; c_s[j] = 0.f; }
    __syncthreads();

    const v4bf* wrow = (const v4bf*)(wsm + j * WPAD);

    for (int t = 0; t < SEQLEN; ++t) {
        const int tt = dir ? (SEQLEN - 1 - t) : t;
        float acc = G[(size_t)tt * GDIM + j];
        if (t + 1 < SEQLEN) {
            const int tn = dir ? (SEQLEN - 2 - t) : (t + 1);
            __builtin_prefetch(&G[(size_t)tn * GDIM + j], 0, 0);
        }
        #pragma unroll 12
        for (int k4 = 0; k4 < HDIM / 4; ++k4) {
            v4bf wv = wrow[k4];                 // ds_load_b64, conflict-free
            const int k = k4 * 4;
            acc += (float)wv[0] * h_s[k]     + (float)wv[1] * h_s[k + 1]
                 + (float)wv[2] * h_s[k + 2] + (float)wv[3] * h_s[k + 3];
        }
        g_s[j] = acc;
        __syncthreads();
        if (j < HDIM) {
            float gi = g_s[j], gfv = g_s[HDIM + j];
            float gg = g_s[2 * HDIM + j], go = g_s[3 * HDIM + j];
            float si = 1.f / (1.f + __expf(-gi));
            float sf = 1.f / (1.f + __expf(-gfv));
            float so = 1.f / (1.f + __expf(-go));
            float cc = sf * c_s[j] + si * tanhf(gg);
            float hh = so * tanhf(cc);
            c_s[j] = cc; h_s[j] = hh;
            hout[(size_t)tt * HDIM + j] = hh;
        }
        __syncthreads();
    }
#endif
}

// ---------------------------------------------------------------------------
// 5) Tag projection: feats[t][tag] = b_tag[tag] + [hf|hb] . w_tag[tag]
// ---------------------------------------------------------------------------
__global__ void feats_kernel(const float* __restrict__ hf, const float* __restrict__ hb,
                             const float* __restrict__ w_tag, const float* __restrict__ b_tag,
                             float* __restrict__ feats) {
#if defined(__HIP_DEVICE_COMPILE__)
    int idx = blockIdx.x * 256 + threadIdx.x;
    if (idx >= SEQLEN * NTAG) return;
    int t = idx / NTAG, tag = idx - t * NTAG;
    const float* wt  = w_tag + tag * (2 * HDIM);
    const float* hfr = hf + (size_t)t * HDIM;
    const float* hbr = hb + (size_t)t * HDIM;
    float acc = b_tag[tag];
    #pragma unroll 8
    for (int k = 0; k < HDIM; ++k) acc += hfr[k] * wt[k];
    #pragma unroll 8
    for (int k = 0; k < HDIM; ++k) acc += hbr[k] * wt[HDIM + k];
    feats[idx] = acc;
#endif
}

// ---------------------------------------------------------------------------
// 6) Viterbi: one wave32, wave-synchronous shuffles; backpointers as bytes in
//    LDS (8192*11 = 88KB); first-max tie-break matches jnp.argmax.
// ---------------------------------------------------------------------------
__global__ void viterbi_kernel(const float* __restrict__ feats,
                               const float* __restrict__ trans,
                               float* __restrict__ out, int out_size) {
#if defined(__HIP_DEVICE_COMPILE__)
    extern __shared__ unsigned char bp_s[];
    const int lane = threadIdx.x;
    float tr[NTAG];                                       // trans[lane(next)][p(prev)]
    #pragma unroll
    for (int p = 0; p < NTAG; ++p)
        tr[p] = (lane < NTAG) ? trans[lane * NTAG + p] : NEGV;

    float v = (lane == START_TAG) ? 0.f : NEGV;
    for (int t = 0; t < SEQLEN; ++t) {
        float best = -3.4e38f; int bp = 0;
        #pragma unroll
        for (int p = 0; p < NTAG; ++p) {
            float vp = __shfl(v, p, 32);
            float s  = vp + tr[p];
            if (s > best) { best = s; bp = p; }
        }
        float nv = best + ((lane < NTAG) ? feats[t * NTAG + lane] : 0.f);
        v = (lane < NTAG) ? nv : NEGV;
        if (lane < NTAG) bp_s[t * NTAG + lane] = (unsigned char)bp;
    }
    float term = v + ((lane < NTAG) ? trans[STOP_TAG * NTAG + lane] : 0.f);
    float best = -3.4e38f; int bt = 0;
    #pragma unroll
    for (int p = 0; p < NTAG; ++p) {
        float tv = __shfl(term, p, 32);
        if (tv > best) { best = tv; bt = p; }
    }
    if (lane == 0) {
        if (out_size > 0) out[0] = best;
        int tag = bt;
        for (int t = SEQLEN - 1; t >= 0; --t) {
            if (1 + t < out_size) out[1 + t] = (float)tag;
            tag = bp_s[t * NTAG + tag];
        }
    }
#endif
}

// ---------------------------------------------------------------------------
extern "C" void kernel_launch(void* const* d_in, const int* in_sizes, int n_in,
                              void* d_out, int out_size, void* d_ws, size_t ws_size,
                              hipStream_t stream) {
    (void)in_sizes; (void)n_in; (void)ws_size;
    const int*   sentence = (const int*)  d_in[0];
    const float* emb      = (const float*)d_in[1];
    const float* w_ih_f   = (const float*)d_in[2];
    const float* w_hh_f   = (const float*)d_in[3];
    const float* b_ih_f   = (const float*)d_in[4];
    const float* b_hh_f   = (const float*)d_in[5];
    const float* w_ih_b   = (const float*)d_in[6];
    const float* w_hh_b   = (const float*)d_in[7];
    const float* b_ih_b   = (const float*)d_in[8];
    const float* b_hh_b   = (const float*)d_in[9];
    const float* w_tag    = (const float*)d_in[10];
    const float* b_tag    = (const float*)d_in[11];
    const float* trans    = (const float*)d_in[12];

    // workspace carve-up (all sizes already 256B-aligned)
    char* ws = (char*)d_ws;
    bf16_t* xbf  = (bf16_t*)ws;  ws += (size_t)SEQLEN * EDIM * 2;     // 6.29 MB
    bf16_t* wfb  = (bf16_t*)ws;  ws += (size_t)GDIM * EDIM * 2;       // 0.59 MB
    bf16_t* wbb  = (bf16_t*)ws;  ws += (size_t)GDIM * EDIM * 2;
    float*  Gf   = (float*)ws;   ws += (size_t)SEQLEN * GDIM * 4;     // 25.2 MB
    float*  Gb   = (float*)ws;   ws += (size_t)SEQLEN * GDIM * 4;
    float*  hf   = (float*)ws;   ws += (size_t)SEQLEN * HDIM * 4;     // 6.29 MB
    float*  hb   = (float*)ws;   ws += (size_t)SEQLEN * HDIM * 4;
    float*  feat = (float*)ws;   ws += (size_t)SEQLEN * NTAG * 4;

    const int LSTM_LDS = GDIM * WPAD * 2 + (HDIM + HDIM + GDIM) * 4;  // 305,664 B (<320KB WGP LDS)
    const int VIT_LDS  = SEQLEN * NTAG;                               // 90,112 B
    (void)hipFuncSetAttribute((const void*)lstm_scan_kernel,
                              hipFuncAttributeMaxDynamicSharedMemorySize, LSTM_LDS);
    (void)hipFuncSetAttribute((const void*)viterbi_kernel,
                              hipFuncAttributeMaxDynamicSharedMemorySize, VIT_LDS);

    embed_bf16_kernel<<<(SEQLEN * EDIM + 255) / 256, 256, 0, stream>>>(sentence, emb, xbf);
    cvt_bf16_kernel<<<(GDIM * EDIM + 255) / 256, 256, 0, stream>>>(w_ih_f, wfb, GDIM * EDIM);
    cvt_bf16_kernel<<<(GDIM * EDIM + 255) / 256, 256, 0, stream>>>(w_ih_b, wbb, GDIM * EDIM);

    const int tiles = (SEQLEN / 16) * (GDIM / 16);                    // 24576, 8 waves/block
    gates_gemm_kernel<<<tiles / 8, 256, 0, stream>>>(xbf, wfb, b_ih_f, b_hh_f, Gf);
    gates_gemm_kernel<<<tiles / 8, 256, 0, stream>>>(xbf, wbb, b_ih_b, b_hh_b, Gb);

    lstm_scan_kernel<<<2, GDIM, LSTM_LDS, stream>>>(w_hh_f, w_hh_b, Gf, Gb, hf, hb);

    feats_kernel<<<(SEQLEN * NTAG + 255) / 256, 256, 0, stream>>>(hf, hb, w_tag, b_tag, feat);
    viterbi_kernel<<<1, 32, VIT_LDS, stream>>>(feat, trans, (float*)d_out, out_size);
}
